// CustomLlamaDecoderLayer_64304250356188
// MI455X (gfx1250) — matrix operand order repaired and checked by reading, not compiled, not run on hardware
//
#include <hip/hip_runtime.h>

// ---------------------------------------------------------------------------
// CDNA5 (gfx1250) Llama decoder layer.  bf16 WMMA everywhere, f32 accumulate.
// Shapes fixed by the reference: B=1, S=2048, D=2048, H=32, HD=64, FF=8192.
// GEMM: 128x128 block tile, double-buffered async LDS staging.
// ---------------------------------------------------------------------------

typedef unsigned short u16;
typedef __attribute__((ext_vector_type(16))) __bf16         v16bf;
typedef __attribute__((ext_vector_type(8)))  float          v8f;
typedef __attribute__((ext_vector_type(16))) unsigned short v16us;
typedef __attribute__((ext_vector_type(8)))  unsigned short v8us;
typedef __attribute__((ext_vector_type(4)))  int            v4i;

static constexpr int S_LEN = 2048;
static constexpr int D_DIM = 2048;
static constexpr int H_CNT = 32;
static constexpr int HD    = 64;
static constexpr int FF    = 8192;

union Frag16 {
  v16bf bf;
  v16us us;
  v8us  h[2];
};

__device__ __forceinline__ u16 f32_to_bf16(float f) {
  unsigned int u = __float_as_uint(f);
  unsigned int r = 0x7FFFu + ((u >> 16) & 1u);
  return (u16)((u + r) >> 16);
}

// ---- CDNA5 async global->LDS copy (ASYNCcnt) with safe fallback -----------
#if defined(__has_builtin)
#if __has_builtin(__builtin_amdgcn_global_load_async_to_lds_b128)
#define HAVE_ASYNC_LDS 1
#endif
#endif
#ifndef HAVE_ASYNC_LDS
#define HAVE_ASYNC_LDS 0
#endif

__device__ __forceinline__ void copy16_async(const u16* g, u16* l) {
#if HAVE_ASYNC_LDS
  __builtin_amdgcn_global_load_async_to_lds_b128(
      (__attribute__((address_space(1))) v4i*)(u16*)g,
      (__attribute__((address_space(3))) v4i*)l, 0, 0);
#else
  *(v8us*)l = *(const v8us*)g;
#endif
}

__device__ __forceinline__ void wait_async_copies() {
#if HAVE_ASYNC_LDS
#if __has_builtin(__builtin_amdgcn_s_wait_asynccnt)
  __builtin_amdgcn_s_wait_asynccnt(0);
#else
  asm volatile("s_wait_asynccnt 0x0" ::: "memory");
#endif
#endif
}

// --------------------------- f32 -> bf16 convert ---------------------------
__global__ void to_bf16_kernel(const float* __restrict__ in, u16* __restrict__ out,
                               size_t n) {
  size_t i = (size_t)blockIdx.x * blockDim.x + threadIdx.x;
  size_t stride = (size_t)gridDim.x * blockDim.x;
  for (; i < n; i += stride) out[i] = f32_to_bf16(in[i]);
}

// --------------------------- RMSNorm -> bf16 -------------------------------
__global__ void __launch_bounds__(256) rmsnorm_bf16_kernel(
    const float* __restrict__ x, const float* __restrict__ w,
    u16* __restrict__ out, int d) {
  __shared__ float red[256];
  int row = blockIdx.x;
  int tid = threadIdx.x;
  const float* xr = x + (size_t)row * d;
  float ss = 0.f;
  for (int i = tid; i < d; i += 256) {
    float v = xr[i];
    ss += v * v;
  }
  red[tid] = ss;
  __syncthreads();
  for (int s = 128; s > 0; s >>= 1) {
    if (tid < s) red[tid] += red[tid + s];
    __syncthreads();
  }
  float inv = rsqrtf(red[0] / (float)d + 1e-5f);
  for (int i = tid; i < d; i += 256)
    out[(size_t)row * d + i] = f32_to_bf16(xr[i] * inv * w[i]);
}

// --------------------------- WMMA GEMM -------------------------------------
// C[M,N] = A[M,K](bf16) @ W[N,K]^T(bf16)  (+ resid), f32 out.
// Block tile 128x128, 256 threads = 8 waves in 2(M)x4(N); each wave 64x32.
// K stepped by 32 through double-buffered LDS tiles (padded stride vs banks).
static constexpr int AST = 40;  // LDS row stride in halves (32 data + 8 pad)

__global__ void __launch_bounds__(256) gemm_bf16_kernel(
    const u16* __restrict__ A, const u16* __restrict__ W,
    const float* __restrict__ resid, float* __restrict__ C,
    int M, int N, int K) {
  __shared__ __align__(16) u16 As[2][128 * AST];
  __shared__ __align__(16) u16 Bs[2][128 * AST];

  int tid  = threadIdx.x;
  int lane = tid & 31;
  int wid  = tid >> 5;
  int half = lane >> 4;
  int l16  = lane & 15;

  int bm = blockIdx.y * 128;
  int bn = blockIdx.x * 128;
  int wm = (wid & 1) * 64;   // local row base within tile (4 m-tiles of 16)
  int wn = (wid >> 1) * 32;  // local col base within tile (2 n-tiles of 16)

  // cooperative fill: each thread moves 32B of A and 32B of B per stage
  int crow = tid >> 1;        // 0..127
  int cseg = (tid & 1) * 16;  // half offset 0 or 16

  const u16* gA = A + (size_t)(bm + crow) * K + cseg;
  const u16* gB = W + (size_t)(bn + crow) * K + cseg;
  u16* lA = &As[0][crow * AST + cseg];
  u16* lB = &Bs[0][crow * AST + cseg];
  const int BUFSTRIDE = 128 * AST;

  auto stage_in = [&](int buf, int k0) {
    copy16_async(gA + k0,     lA + buf * BUFSTRIDE);
    copy16_async(gA + k0 + 8, lA + buf * BUFSTRIDE + 8);
    copy16_async(gB + k0,     lB + buf * BUFSTRIDE);
    copy16_async(gB + k0 + 8, lB + buf * BUFSTRIDE + 8);
  };

  v8f acc[4][2] = {};

  stage_in(0, 0);
  wait_async_copies();
  __syncthreads();

  int buf = 0;
  for (int k0 = 0; k0 < K; k0 += 32) {
    int nk = k0 + 32;
    if (nk < K) stage_in(buf ^ 1, nk);

    const u16* Abase = &As[buf][0];
    const u16* Bbase = &Bs[buf][0];
    Frag16 a[4], b[2];
    // A fragment (16x32): lane -> row=l16, K = half*16 .. +15 (contiguous).
    #pragma unroll
    for (int mi = 0; mi < 4; ++mi) {
      const u16* ap = Abase + (wm + mi * 16 + l16) * AST + half * 16;
      a[mi].h[0] = *(const v8us*)ap;
      a[mi].h[1] = *(const v8us*)(ap + 8);
    }
    // B fragment (32x16): lane -> col=l16, K = half*16 .. +15 (contiguous).
    #pragma unroll
    for (int ni = 0; ni < 2; ++ni) {
      const u16* bp = Bbase + (wn + ni * 16 + l16) * AST + half * 16;
      b[ni].us = *(const v16us*)bp;
    }
    #pragma unroll
    for (int mi = 0; mi < 4; ++mi)
      #pragma unroll
      for (int ni = 0; ni < 2; ++ni)
        acc[mi][ni] = __builtin_amdgcn_wmma_f32_16x16x32_bf16(
            false, a[mi].bf, false, b[ni].bf, (short)0, acc[mi][ni], false, false);

    wait_async_copies();
    __syncthreads();
    buf ^= 1;
  }

  const bool has_res = (resid != nullptr);
  #pragma unroll
  for (int mi = 0; mi < 4; ++mi)
    #pragma unroll
    for (int ni = 0; ni < 2; ++ni)
      #pragma unroll
      for (int r = 0; r < 8; ++r) {
        int row = bm + wm + mi * 16 + half * 8 + r;
        int col = bn + wn + ni * 16 + l16;
        size_t idx = (size_t)row * N + col;
        float v = acc[mi][ni][r];
        if (has_res) v += resid[idx];
        C[idx] = v;
      }
}

// --------------------------- RoPE (Q scaled) -------------------------------
// In:  qf/kf f32 [S, D] ; Out: qb/kb bf16 head-major [H, S, HD].
__global__ void rope_qk_kernel(const float* __restrict__ qf,
                               const float* __restrict__ kf,
                               u16* __restrict__ qb, u16* __restrict__ kb) {
  int idx = blockIdx.x * blockDim.x + threadIdx.x;  // S*H*(HD/2)
  int d2 = idx & 31;
  int h  = (idx >> 5) & 31;
  int s  = idx >> 10;
  if (s >= S_LEN) return;
  float inv_freq = __powf(10000.0f, -(float)(2 * d2) / (float)HD);
  float ang = (float)s * inv_freq;
  float c = __cosf(ang), sn = __sinf(ang);
  size_t in0  = (size_t)s * D_DIM + h * HD + d2;
  size_t out0 = ((size_t)h * S_LEN + s) * HD + d2;
  const float scale = 0.125f;  // HD^-0.5 folded into Q
  float q1 = qf[in0], q2 = qf[in0 + 32];
  qb[out0]      = f32_to_bf16((q1 * c - q2 * sn) * scale);
  qb[out0 + 32] = f32_to_bf16((q2 * c + q1 * sn) * scale);
  float k1 = kf[in0], k2 = kf[in0 + 32];
  kb[out0]      = f32_to_bf16(k1 * c - k2 * sn);
  kb[out0 + 32] = f32_to_bf16(k2 * c + k1 * sn);
}

// --------------------------- V transpose -----------------------------------
// vf f32 [S, D] -> vt bf16 [H, HD, S]
__global__ void v_transpose_kernel(const float* __restrict__ vf,
                                   u16* __restrict__ vt) {
  int idx = blockIdx.x * blockDim.x + threadIdx.x;  // S*D
  if (idx >= S_LEN * D_DIM) return;
  int s = idx >> 11;     // / D
  int rem = idx & 2047;
  int h = rem >> 6;      // / HD
  int hd = rem & 63;
  vt[((size_t)h * HD + hd) * S_LEN + s] = f32_to_bf16(vf[idx]);
}

// --------------------------- Flash attention -------------------------------
// One wave = one (head, 16-query tile).  Online softmax, WMMA QK^T and PV.
__global__ void __launch_bounds__(256) attention_kernel(
    const u16* __restrict__ qb,   // [H, S, HD]
    const u16* __restrict__ kb,   // [H, S, HD]
    const u16* __restrict__ vt,   // [H, HD, S]
    u16* __restrict__ attn) {     // [S, D] bf16
  __shared__ __align__(16) u16 pbuf[8][16 * 32];
  int lane = threadIdx.x & 31;
  int wid  = threadIdx.x >> 5;
  int half = lane >> 4;
  int l16  = lane & 15;
  int t    = blockIdx.x * 8 + wid;
  int head = t >> 7;  // / (S/16) = 128
  int qt   = t & 127;

  // Q A-fragments for HD split into two K=32 chunks.
  Frag16 qa[2];
  #pragma unroll
  for (int c = 0; c < 2; ++c) {
    const u16* qp = qb + ((size_t)head * S_LEN + qt * 16 + l16) * HD + c * 32 + half * 16;
    qa[c].h[0] = *(const v8us*)qp;
    qa[c].h[1] = *(const v8us*)(qp + 8);
  }

  float mrun[8], lrun[8];
  #pragma unroll
  for (int r = 0; r < 8; ++r) { mrun[r] = -1e30f; lrun[r] = 0.f; }
  v8f o[4] = {};

  for (int kc = 0; kc < S_LEN; kc += 32) {
    // ---- scores: two 16x16 C tiles over 32 keys -------------------------
    v8f sc[2];
    #pragma unroll
    for (int nt = 0; nt < 2; ++nt) {
      v8f z = {};
      #pragma unroll
      for (int c = 0; c < 2; ++c) {
        Frag16 bk;
        const u16* kp = kb + ((size_t)head * S_LEN + kc + nt * 16 + l16) * HD +
                        c * 32 + half * 16;
        bk.us = *(const v16us*)kp;
        z = __builtin_amdgcn_wmma_f32_16x16x32_bf16(
            false, qa[c].bf, false, bk.bf, (short)0, z, false, false);
      }
      sc[nt] = z;
    }

    // ---- online softmax over the 32-key chunk ---------------------------
    #pragma unroll
    for (int r = 0; r < 8; ++r) {
      float mx = fmaxf(sc[0][r], sc[1][r]);
      #pragma unroll
      for (int m = 1; m < 16; m <<= 1) mx = fmaxf(mx, __shfl_xor(mx, m, 32));
      float mn = fmaxf(mrun[r], mx);
      float corr = __expf(mrun[r] - mn);
      float p0 = __expf(sc[0][r] - mn);
      float p1 = __expf(sc[1][r] - mn);
      float rs = p0 + p1;
      #pragma unroll
      for (int m = 1; m < 16; m <<= 1) rs += __shfl_xor(rs, m, 32);
      lrun[r] = lrun[r] * corr + rs;
      mrun[r] = mn;
      #pragma unroll
      for (int ni = 0; ni < 4; ++ni) o[ni][r] *= corr;
      // stage P (C layout) into LDS for relayout to an A-fragment
      int row = half * 8 + r;
      pbuf[wid][row * 32 + l16]      = f32_to_bf16(p0);
      pbuf[wid][row * 32 + 16 + l16] = f32_to_bf16(p1);
    }
    asm volatile("s_wait_dscnt 0x0" ::: "memory");

    // ---- P as A-fragment (16x32) ----------------------------------------
    Frag16 pf;
    {
      const u16* pp = &pbuf[wid][l16 * 32 + half * 16];
      pf.h[0] = *(const v8us*)pp;
      pf.h[1] = *(const v8us*)(pp + 8);
    }

    // ---- PV: O(16x64) += P(16x32) x V(32x64) ----------------------------
    #pragma unroll
    for (int ni = 0; ni < 4; ++ni) {
      Frag16 bv;
      const u16* vp = vt + ((size_t)head * HD + ni * 16 + l16) * S_LEN +
                      kc + half * 16;
      bv.us = *(const v16us*)vp;
      o[ni] = __builtin_amdgcn_wmma_f32_16x16x32_bf16(
          false, pf.bf, false, bv.bf, (short)0, o[ni], false, false);
    }
    asm volatile("s_wait_dscnt 0x0" ::: "memory");  // WAR on pbuf
  }

  // ---- epilogue: normalize, write bf16 [S, D] ---------------------------
  #pragma unroll
  for (int r = 0; r < 8; ++r) {
    float inv = 1.0f / lrun[r];
    int row = qt * 16 + half * 8 + r;
    #pragma unroll
    for (int ni = 0; ni < 4; ++ni) {
      int col = head * HD + ni * 16 + l16;
      attn[(size_t)row * D_DIM + col] = f32_to_bf16(o[ni][r] * inv);
    }
  }
}

// --------------------------- gate * silu(up) -> bf16 -----------------------
__global__ void gate_silu_kernel(const float* __restrict__ g,
                                 const float* __restrict__ u,
                                 u16* __restrict__ act, size_t n) {
  size_t i = (size_t)blockIdx.x * blockDim.x + threadIdx.x;
  size_t stride = (size_t)gridDim.x * blockDim.x;
  for (; i < n; i += stride) {
    float uu = u[i];
    float sv = uu / (1.f + __expf(-uu));
    act[i] = f32_to_bf16(g[i] * sv);
  }
}

// ===========================================================================
extern "C" void kernel_launch(void* const* d_in, const int* in_sizes, int n_in,
                              void* d_out, int out_size, void* d_ws, size_t ws_size,
                              hipStream_t stream) {
  (void)in_sizes; (void)n_in; (void)out_size; (void)ws_size;
  const float* x      = (const float*)d_in[0];
  const float* w_in   = (const float*)d_in[1];
  const float* wq     = (const float*)d_in[2];
  const float* wk     = (const float*)d_in[3];
  const float* wv     = (const float*)d_in[4];
  const float* wo     = (const float*)d_in[5];
  const float* w_post = (const float*)d_in[6];
  const float* wg     = (const float*)d_in[7];
  const float* wu     = (const float*)d_in[8];
  const float* wd     = (const float*)d_in[9];
  float* out = (float*)d_out;

  char* ws = (char*)d_ws;
  size_t off = 0;
  auto alloc = [&](size_t bytes) -> char* {
    off = (off + 255) & ~(size_t)255;
    char* p = ws + off;
    off += bytes;
    return p;
  };

  const size_t SD = (size_t)S_LEN * D_DIM;
  const size_t SF = (size_t)S_LEN * FF;
  const size_t DD = (size_t)D_DIM * D_DIM;
  const size_t DF = (size_t)D_DIM * FF;

  u16* wq_b = (u16*)alloc(DD * 2);
  u16* wk_b = (u16*)alloc(DD * 2);
  u16* wv_b = (u16*)alloc(DD * 2);
  u16* wo_b = (u16*)alloc(DD * 2);
  u16* wg_b = (u16*)alloc(DF * 2);
  u16* wu_b = (u16*)alloc(DF * 2);
  u16* wd_b = (u16*)alloc(DF * 2);
  u16* h1   = (u16*)alloc(SD * 2);
  float* qf = (float*)alloc(SD * 4);
  float* kf = (float*)alloc(SD * 4);
  float* vf = (float*)alloc(SD * 4);
  u16* qb   = (u16*)alloc(SD * 2);
  u16* kb   = (u16*)alloc(SD * 2);
  u16* vt   = (u16*)alloc(SD * 2);
  u16* attn = (u16*)alloc(SD * 2);
  float* x1 = (float*)alloc(SD * 4);
  u16* h2   = (u16*)alloc(SD * 2);
  float* gf = (float*)alloc(SF * 4);
  float* uf = (float*)alloc(SF * 4);
  u16* act  = (u16*)alloc(SF * 2);

  auto cvt = [&](const float* src, u16* dst, size_t n) {
    int blocks = (int)((n + 2047) / 2048);
    to_bf16_kernel<<<blocks, 256, 0, stream>>>(src, dst, n);
  };
  auto gemm = [&](const u16* A, const u16* W, const float* resid, float* C,
                  int M, int N, int K) {
    dim3 grid(N / 128, M / 128);
    gemm_bf16_kernel<<<grid, 256, 0, stream>>>(A, W, resid, C, M, N, K);
  };

  // 1) weight conversions
  cvt(wq, wq_b, DD); cvt(wk, wk_b, DD); cvt(wv, wv_b, DD); cvt(wo, wo_b, DD);
  cvt(wg, wg_b, DF); cvt(wu, wu_b, DF); cvt(wd, wd_b, DF);

  // 2) input RMSNorm
  rmsnorm_bf16_kernel<<<S_LEN, 256, 0, stream>>>(x, w_in, h1, D_DIM);

  // 3) QKV projections
  gemm(h1, wq_b, nullptr, qf, S_LEN, D_DIM, D_DIM);
  gemm(h1, wk_b, nullptr, kf, S_LEN, D_DIM, D_DIM);
  gemm(h1, wv_b, nullptr, vf, S_LEN, D_DIM, D_DIM);

  // 4) RoPE + layout changes
  {
    int threads = S_LEN * H_CNT * (HD / 2);
    rope_qk_kernel<<<threads / 256, 256, 0, stream>>>(qf, kf, qb, kb);
    v_transpose_kernel<<<(int)(SD / 256), 256, 0, stream>>>(vf, vt);
  }

  // 5) attention (flash, per-head)
  attention_kernel<<<(H_CNT * (S_LEN / 16)) / 8, 256, 0, stream>>>(qb, kb, vt, attn);

  // 6) O projection + residual
  gemm(attn, wo_b, x, x1, S_LEN, D_DIM, D_DIM);

  // 7) post norm
  rmsnorm_bf16_kernel<<<S_LEN, 256, 0, stream>>>(x1, w_post, h2, D_DIM);

  // 8) MLP
  gemm(h2, wg_b, nullptr, gf, S_LEN, FF, D_DIM);
  gemm(h2, wu_b, nullptr, uf, S_LEN, FF, D_DIM);
  gate_silu_kernel<<<(int)((SF + 2047) / 2048), 256, 0, stream>>>(gf, uf, act, SF);
  gemm(act, wd_b, x1, out, S_LEN, D_DIM, FF);
}
